// Windows_Attention_90340342104422
// MI455X (gfx1250) — matrix operand order repaired and checked by reading, not compiled, not run on hardware
//
#include <hip/hip_runtime.h>
#include <hip/hip_bf16.h>

// ---------------------------------------------------------------------------
// SwinV2 window attention, fully fused per-window, bf16 WMMA on gfx1250.
//   x: (8192, 64, 128) f32   -> out: (8192, 64, 128) f32
// One block per window, 256 threads = 8 wave32 waves.
// ---------------------------------------------------------------------------

typedef __bf16  bf16;
typedef __bf16  v16bf __attribute__((ext_vector_type(16)));
typedef float   v8f   __attribute__((ext_vector_type(8)));

#define NTOK   64      // tokens per window
#define CDIM   128     // embed dim
#define NHEAD  4
#define HD     32      // head dim
#define NWIN   64      // mask groups
#define NBLK   8192

// LDS region layout (bytes). Regions are reused across phases (barriers between).
//  R1 (69632 B): xb (64 x 136 bf16) -> scores (4 x 64 x 68 f32) -> ob (64 x 136 bf16)
//  R2 (36864 B): qb+kb (2 x 64 x 144 bf16) -> probs (4 x 64 x 72 bf16)
//  R3 (18432 B): vT (128 x 72 bf16)
#define R1_OFF 0
#define R2_OFF 69632
#define R3_OFF (69632 + 36864)
#define SMEM_BYTES (69632 + 36864 + 18432)

#define XB_S   136   // bf16 row stride of xb / ob
#define QK_S   144   // bf16 row stride of qb / kb
#define SC_S   68    // f32 row stride of scores
#define PB_S   72    // bf16 row stride of probs
#define VT_S   72    // bf16 row stride of vT

// A operand (16x32 bf16): lane group sel=lane>>4 holds contiguous K in [sel*16, sel*16+16).
__device__ __forceinline__ v16bf load_a16(const bf16* rowbase, int kbase, int sel) {
    const bf16* p = rowbase + kbase + sel * 16;
    union { v16bf v; uint4 q[2]; } u;
    u.q[0] = *(const uint4*)(p);
    u.q[1] = *(const uint4*)(p + 8);
    return u.v;
}
// B operand (32x16 bf16): lane = column, contiguous K in [sel*16, sel*16+16).
__device__ __forceinline__ v16bf load_b16(const bf16* colbase, int kbase, int sel) {
    const bf16* p = colbase + kbase + sel * 16;
    union { v16bf v; uint4 q[2]; } u;
    u.q[0] = *(const uint4*)(p);
    u.q[1] = *(const uint4*)(p + 8);
    return u.v;
}
__device__ __forceinline__ v8f wmma_bf16(v16bf a, v16bf b, v8f c) {
    return __builtin_amdgcn_wmma_f32_16x16x32_bf16(false, a, false, b, (short)0, c,
                                                   false, false);
}

union PK4 { bf16 h[4]; uint2 u; };   // 4 packed bf16
union PK8 { bf16 h[8]; uint4 q; };   // 8 packed bf16

// ---------------- pre-kernels ------------------------------------------------

__global__ void k_convw(const float* __restrict__ qkv_w, const float* __restrict__ proj_w,
                        bf16* __restrict__ wq, bf16* __restrict__ wp) {
    int i = blockIdx.x * blockDim.x + threadIdx.x;
    if (i < 3 * CDIM * CDIM) wq[i] = (bf16)qkv_w[i];
    if (i < CDIM * CDIM)     wp[i] = (bf16)proj_w[i];
}

// CPB MLP: 2 -> 512 (leaky relu 0.01) -> 4, per table position. One block/pos.
__global__ void k_cpb(const float* __restrict__ ct, const float* __restrict__ w1,
                      const float* __restrict__ b1, const float* __restrict__ w2,
                      float* __restrict__ bt) {
    int p = blockIdx.x;        // 0..224
    int j = threadIdx.x;       // 0..127
    float c0 = ct[p * 2 + 0], c1 = ct[p * 2 + 1];
    float acc[NHEAD] = {0.f, 0.f, 0.f, 0.f};
    for (int jj = j; jj < 512; jj += 128) {
        float t = c0 * w1[jj * 2] + c1 * w1[jj * 2 + 1] + b1[jj];
        t = (t >= 0.f) ? t : 0.01f * t;
        #pragma unroll
        for (int h = 0; h < NHEAD; ++h) acc[h] += t * w2[h * 512 + jj];
    }
    __shared__ float red[128][NHEAD];
    #pragma unroll
    for (int h = 0; h < NHEAD; ++h) red[j][h] = acc[h];
    __syncthreads();
    for (int off = 64; off > 0; off >>= 1) {
        if (j < off) {
            #pragma unroll
            for (int h = 0; h < NHEAD; ++h) red[j][h] += red[j + off][h];
        }
        __syncthreads();
    }
    if (j == 0) {
        #pragma unroll
        for (int h = 0; h < NHEAD; ++h) bt[p * NHEAD + h] = red[0][h];
    }
}

// rpb[h][n][m] = 14 * sigmoid(bias_table[rel_pos_index[n][m]][h])
__global__ void k_rpb(const int* __restrict__ rpi, const float* __restrict__ bt,
                      float* __restrict__ rpb) {
    int t = blockIdx.x * blockDim.x + threadIdx.x;   // 16384
    int h = t >> 12;
    int nm = t & 4095;
    float v = bt[rpi[nm] * NHEAD + h];
    rpb[t] = 14.f / (1.f + __expf(-v));
}

// ---------------- main fused attention kernel --------------------------------

__global__ void __launch_bounds__(256)
k_attn(const float* __restrict__ x, const float* __restrict__ mask,
       const float* __restrict__ qkv_b, const float* __restrict__ proj_b,
       const float* __restrict__ logit_scale,
       const bf16* __restrict__ wqkv, const bf16* __restrict__ wproj,
       const float* __restrict__ rpb, float* __restrict__ out) {
    extern __shared__ char smem[];
    bf16*  xb = (bf16*) (smem + R1_OFF);
    float* sc = (float*)(smem + R1_OFF);
    bf16*  ob = (bf16*) (smem + R1_OFF);
    bf16*  qb = (bf16*) (smem + R2_OFF);
    bf16*  kb = (bf16*) (smem + R2_OFF + NTOK * QK_S * 2);
    bf16*  pb = (bf16*) (smem + R2_OFF);
    bf16*  vt = (bf16*) (smem + R3_OFF);

    const int b    = blockIdx.x;
    const int tid  = threadIdx.x;
    // wave id is uniform within a wave: force into an SGPR so all tile-loop
    // control flow is scalar and EXEC stays all-ones around every WMMA.
    const int wave = __builtin_amdgcn_readfirstlane(tid >> 5);
    const int lane = tid & 31;
    const int l16  = lane & 15;
    const int sel  = lane >> 4;

    // ---- P0: load x window, convert to bf16 in LDS -------------------------
    {
        const float4* xg = (const float4*)(x + (size_t)b * NTOK * CDIM);
        for (int i = tid; i < NTOK * CDIM / 4; i += 256) {
            float4 f = xg[i];
            int row = i >> 5;             // 32 float4 per row
            int c   = (i & 31) * 4;
            PK4 t;
            t.h[0] = (bf16)f.x; t.h[1] = (bf16)f.y;
            t.h[2] = (bf16)f.z; t.h[3] = (bf16)f.w;
            *(uint2*)(xb + row * XB_S + c) = t.u;
        }
    }
    __syncthreads();

    // ---- P1: qkv = x @ W^T + b ; q,k row-major bf16, v transposed ----------
    for (int tile = wave; tile < 96; tile += 8) {       // 4 M-tiles x 24 N-tiles
        int mt = tile & 3, nt = tile >> 2;
        v8f acc = {};
        const bf16* ab = xb + (mt * 16 + l16) * XB_S;
        const bf16* bb = wqkv + (size_t)(nt * 16 + l16) * CDIM;
        #pragma unroll
        for (int kc = 0; kc < 4; ++kc)
            acc = wmma_bf16(load_a16(ab, kc * 32, sel), load_b16(bb, kc * 32, sel), acc);
        int col = nt * 16 + l16;
        float bias = qkv_b[col];
        if (nt < 8) {                                   // q
            #pragma unroll
            for (int r = 0; r < 8; ++r)
                qb[(mt * 16 + r + sel * 8) * QK_S + col] = (bf16)(acc[r] + bias);
        } else if (nt < 16) {                           // k
            #pragma unroll
            for (int r = 0; r < 8; ++r)
                kb[(mt * 16 + r + sel * 8) * QK_S + (col - CDIM)] = (bf16)(acc[r] + bias);
        } else {                                        // v, store transposed
            int d = col - 2 * CDIM;
            #pragma unroll
            for (int r = 0; r < 8; ++r)
                vt[d * VT_S + (mt * 16 + r + sel * 8)] = (bf16)(acc[r] + bias);
        }
    }
    __syncthreads();

    // ---- P2: L2-normalize q,k per (row, head); fold logit scale into q -----
    for (int s = tid; s < 512; s += 256) {
        int m  = s >> 8;                 // 0 = q, 1 = k
        int rr = (s >> 2) & 63;
        int h  = s & 3;
        bf16* base = (m ? kb : qb) + rr * QK_S + h * HD;
        PK8 u[4];
        #pragma unroll
        for (int j = 0; j < 4; ++j) u[j].q = *(const uint4*)(base + j * 8);
        float ss = 0.f;
        #pragma unroll
        for (int j = 0; j < 4; ++j)
            #pragma unroll
            for (int i = 0; i < 8; ++i) { float v = (float)u[j].h[i]; ss += v * v; }
        float inv = 1.f / fmaxf(sqrtf(ss), 1e-12f);
        if (m == 0) inv *= __expf(fminf(logit_scale[h], 4.6051702f));  // ln(100)
        #pragma unroll
        for (int j = 0; j < 4; ++j) {
            #pragma unroll
            for (int i = 0; i < 8; ++i) u[j].h[i] = (bf16)((float)u[j].h[i] * inv);
            *(uint4*)(base + j * 8) = u[j].q;
        }
    }
    __syncthreads();

    // ---- P3: scores = qn @ kn^T per head (f32 to LDS) ----------------------
    for (int tile = wave; tile < 64; tile += 8) {       // 4h x 4M x 4N
        int h = tile >> 4, mt = (tile >> 2) & 3, nt = tile & 3;
        v8f acc = {};
        acc = wmma_bf16(load_a16(qb + (mt * 16 + l16) * QK_S + h * HD, 0, sel),
                        load_b16(kb + (nt * 16 + l16) * QK_S + h * HD, 0, sel), acc);
        int col = nt * 16 + l16;
        #pragma unroll
        for (int r = 0; r < 8; ++r)
            sc[h * (NTOK * SC_S) + (mt * 16 + r + sel * 8) * SC_S + col] = acc[r];
    }
    __syncthreads();

    // ---- P4: + rpb + mask, softmax over last dim; bf16 probs ---------------
    {
        int h = tid >> 6, row = tid & 63;
        const float4* r4 = (const float4*)(rpb  + h * (NTOK * NTOK) + row * NTOK);
        const float4* m4 = (const float4*)(mask + (size_t)(b & (NWIN - 1)) * NTOK * NTOK
                                                + row * NTOK);
        float4* s4 = (float4*)(sc + h * (NTOK * SC_S) + row * SC_S);
        float mx = -1e30f;
        for (int i = 0; i < NTOK / 4; ++i) {
            float4 s = s4[i], r = r4[i], mm = m4[i];
            s.x += r.x + mm.x; s.y += r.y + mm.y;
            s.z += r.z + mm.z; s.w += r.w + mm.w;
            s4[i] = s;
            mx = fmaxf(mx, fmaxf(fmaxf(s.x, s.y), fmaxf(s.z, s.w)));
        }
        float sum = 0.f;
        for (int i = 0; i < NTOK / 4; ++i) {
            float4 s = s4[i];
            s.x = __expf(s.x - mx); s.y = __expf(s.y - mx);
            s.z = __expf(s.z - mx); s.w = __expf(s.w - mx);
            s4[i] = s;
            sum += (s.x + s.y) + (s.z + s.w);
        }
        float inv = 1.f / sum;
        bf16* prow = pb + h * (NTOK * PB_S) + row * PB_S;
        for (int i = 0; i < NTOK / 4; ++i) {
            float4 s = s4[i];
            PK4 t;
            t.h[0] = (bf16)(s.x * inv); t.h[1] = (bf16)(s.y * inv);
            t.h[2] = (bf16)(s.z * inv); t.h[3] = (bf16)(s.w * inv);
            *(uint2*)(prow + i * 4) = t.u;
        }
    }
    __syncthreads();

    // ---- P5: o = probs @ v (bf16 to LDS, token-major, C-channel cols) ------
    for (int tile = wave; tile < 32; tile += 8) {       // 4h x 4M x 2N
        int h = tile >> 3, mt = (tile >> 1) & 3, nt = tile & 1;
        v8f acc = {};
        const bf16* ab = pb + h * (NTOK * PB_S) + (mt * 16 + l16) * PB_S;
        const bf16* bb = vt + (h * HD + nt * 16 + l16) * VT_S;
        #pragma unroll
        for (int kc = 0; kc < 2; ++kc)
            acc = wmma_bf16(load_a16(ab, kc * 32, sel), load_b16(bb, kc * 32, sel), acc);
        int col = h * HD + nt * 16 + l16;
        #pragma unroll
        for (int r = 0; r < 8; ++r)
            ob[(mt * 16 + r + sel * 8) * XB_S + col] = (bf16)acc[r];
    }
    __syncthreads();

    // ---- P6: out = o @ proj_w^T + proj_b (f32 to global) -------------------
    for (int tile = wave; tile < 32; tile += 8) {       // 4M x 8N
        int mt = tile >> 3, nt = tile & 7;
        v8f acc = {};
        const bf16* ab = ob + (mt * 16 + l16) * XB_S;
        const bf16* bb = wproj + (size_t)(nt * 16 + l16) * CDIM;
        #pragma unroll
        for (int kc = 0; kc < 4; ++kc)
            acc = wmma_bf16(load_a16(ab, kc * 32, sel), load_b16(bb, kc * 32, sel), acc);
        int col = nt * 16 + l16;
        float bias = proj_b[col];
        float* og = out + (size_t)b * NTOK * CDIM;
        #pragma unroll
        for (int r = 0; r < 8; ++r)
            og[(mt * 16 + r + sel * 8) * CDIM + col] = acc[r] + bias;
    }
}

// ---------------- launch -----------------------------------------------------

extern "C" void kernel_launch(void* const* d_in, const int* in_sizes, int n_in,
                              void* d_out, int out_size, void* d_ws, size_t ws_size,
                              hipStream_t stream) {
    const float* x      = (const float*)d_in[0];
    const float* mask   = (const float*)d_in[1];
    const float* qkv_w  = (const float*)d_in[2];
    const float* qkv_b  = (const float*)d_in[3];
    const float* proj_w = (const float*)d_in[4];
    const float* proj_b = (const float*)d_in[5];
    const float* lscale = (const float*)d_in[6];
    const float* cpb_w1 = (const float*)d_in[7];
    const float* cpb_b1 = (const float*)d_in[8];
    const float* cpb_w2 = (const float*)d_in[9];
    const float* ctable = (const float*)d_in[10];
    const int*   rpi    = (const int*)  d_in[11];
    float* out = (float*)d_out;

    // workspace layout (all 16B aligned)
    char* ws = (char*)d_ws;
    bf16*  wq_bf  = (bf16*) (ws + 0);        // 49152 bf16 = 98304 B
    bf16*  wp_bf  = (bf16*) (ws + 98304);    // 16384 bf16 = 32768 B
    float* btab   = (float*)(ws + 131072);   // 225*4 f32
    float* rpbtab = (float*)(ws + 135168);   // 4*64*64 f32

    k_convw<<<(3 * CDIM * CDIM + 255) / 256, 256, 0, stream>>>(qkv_w, proj_w, wq_bf, wp_bf);
    k_cpb<<<225, 128, 0, stream>>>(ctable, cpb_w1, cpb_b1, cpb_w2, btab);
    k_rpb<<<(NHEAD * NTOK * NTOK) / 256, 256, 0, stream>>>(rpi, btab, rpbtab);
    k_attn<<<NBLK, 256, SMEM_BYTES, stream>>>(x, mask, qkv_b, proj_b, lscale,
                                              wq_bf, wp_bf, rpbtab, out);
    (void)in_sizes; (void)n_in; (void)out_size; (void)ws_size;
}
